// ChannelEstimator_23871428231185
// MI455X (gfx1250) — matrix operand order repaired and checked by reading, not compiled, not run on hardware
//
#include <hip/hip_runtime.h>
#include <math.h>

typedef __attribute__((ext_vector_type(16))) _Float16 v16h;
typedef __attribute__((ext_vector_type(8)))  _Float16 v8h;
typedef __attribute__((ext_vector_type(8)))  float    v8f;

#define NFFT 64
#define NPIL 8
#define HID  128
#define WAVES 8
#define TILE_B (WAVES*16)

// ---------------- prep kernel: build T = Circulant(window) @ Interp (64x8 complex) ----------------
__global__ void ce_prep(const int* __restrict__ pilot_pos,
                        const float* __restrict__ decay_param,
                        const float* __restrict__ window_logits,
                        float* __restrict__ T) {
  __shared__ float interp[64][8];
  __shared__ float mr[64], mi[64];
  __shared__ float win[8], pil[8];
  int t = threadIdx.x; // 64 threads
  if (t < 8) {
    win[t] = 1.0f / (1.0f + expf(-window_logits[t]));
    pil[t] = (float)pilot_pos[t];
  }
  __syncthreads();
  {
    float fi = (float)t;
    int cnt = 0;
    for (int p = 0; p < 8; ++p) cnt += (pil[p] <= fi) ? 1 : 0;
    int left = cnt - 1; if (left < 0) left = 0; if (left > 6) left = 6;
    int right = left + 1;
    float decay = log1pf(expf(decay_param[0]));   // softplus
    float wl = expf(-decay * fabsf(fi - pil[left]));
    float wr = expf(-decay * fabsf(pil[right] - fi));
    float ws = wl + wr + 1e-12f;
    for (int p = 0; p < 8; ++p) interp[t][p] = 0.0f;
    interp[t][left]  += wl / ws;
    interp[t][right] += wr / ws;
    // circulant taps m[d] = (1/64) sum_{n<8} win[n] e^{2pi i n d/64}
    const float twopi = 6.283185307179586f;
    float sr = 0.0f, si = 0.0f;
    for (int n = 0; n < 8; ++n) {
      float ang = twopi * (float)n * (float)t * (1.0f/64.0f);
      sr += win[n] * cosf(ang);
      si += win[n] * sinf(ang);
    }
    mr[t] = sr * (1.0f/64.0f);
    mi[t] = si * (1.0f/64.0f);
  }
  __syncthreads();
  int k = t;
  for (int p = 0; p < 8; ++p) {
    float tr = 0.0f, ti = 0.0f;
    for (int l = 0; l < 64; ++l) {
      int d = (l - k) & 63;
      float w = interp[l][p];
      tr += mr[d] * w;
      ti += mi[d] * w;
    }
    T[(k*8+p)*2+0] = tr;
    T[(k*8+p)*2+1] = ti;
  }
}

// ---------------- fragment helpers (CDNA5 16x16x32 f16 WMMA layouts) ----------------
__device__ __forceinline__ v16h cat16(v8h lo, v8h hi) {
  return __builtin_shufflevector(lo, hi, 0,1,2,3,4,5,6,7,8,9,10,11,12,13,14,15);
}
// A (16x32, MxK) from row-major [M][rs] halves: lane group g holds K in [8g,8g+8) U [16+8g,16+8g+8)
__device__ __forceinline__ v16h load_A(const _Float16* X, int rs, int lane, int kbase) {
  int g = lane >> 4;
  int m = lane & 15;
  const _Float16* p0 = X + m*rs + kbase + 8*g;
  v8h lo = *(const v8h*)p0;
  v8h hi = *(const v8h*)(p0 + 16);
  return cat16(lo, hi);
}
// B (32x16, KxN) from [N][rs] halves: lane holds N = lane&15, K in [8g,8g+8) U [16+8g,16+8g+8)
__device__ __forceinline__ v16h load_B(const _Float16* WT, int rs, int lane, int ntbase, int kbase) {
  int g = lane >> 4;
  int n = (lane & 15) + ntbase;
  const _Float16* p0 = WT + n*rs + kbase + 8*g;
  v8h lo = *(const v8h*)p0;
  v8h hi = *(const v8h*)(p0 + 16);
  return cat16(lo, hi);
}
// D (16x16 f32): lane L, VGPR r -> [M = r + 8*(L/16)][N = L%16]
__device__ __forceinline__ void store_relu(_Float16* H, int rs, int lane, int ntbase,
                                           v8f acc, const float* bias) {
  int g = lane >> 4;
  int n = lane & 15;
  float bb = bias[ntbase + n];
#pragma unroll
  for (int r = 0; r < 8; ++r) {
    float v = fmaxf(acc[r] + bb, 0.0f);
    H[(r + 8*g)*rs + ntbase + n] = (_Float16)v;
  }
}

// ---------------- main kernel ----------------
__global__ __launch_bounds__(256) void ce_main(
    const float* __restrict__ Y, const float* __restrict__ Xp,
    const int* __restrict__ pilot_pos,
    const float* __restrict__ W1, const float* __restrict__ b1,
    const float* __restrict__ W2, const float* __restrict__ b2,
    const float* __restrict__ W3, const float* __restrict__ b3,
    const float* __restrict__ est_w, const float* __restrict__ alpha_p,
    const float* __restrict__ Tg, float* __restrict__ out, int B)
{
  extern __shared__ __attribute__((aligned(16))) char smem[];
  _Float16* W1T = (_Float16*)smem;          // [128][64]   16KB
  _Float16* W2T = W1T + 128*64;             // [128][128]  32KB
  _Float16* W3T = W2T + 128*128;            // [16][128]   4KB (only out cols 0..15 needed)
  float* b1s = (float*)(W3T + 16*128);
  float* b2s = b1s + 128;
  float* b3s = b2s + 128;
  float* Ts  = b3s + 16;                    // [64][8][2]  4KB
  _Float16* XH2 = (_Float16*)(Ts + 1024);   // per-wave [16][128]: MLP input, later reused as h2
  _Float16* H1  = XH2 + WAVES*16*128;       // per-wave [16][128]
  float* WLS  = (float*)(H1 + WAVES*16*128);// per-wave [16][8][2]
  float* COMB = WLS + WAVES*16*8*2;         // per-wave [16][16]

  const int t = threadIdx.x;
  // cooperative weight conversion f32 -> f16, transposed to [N][K]
  for (int i = t; i < 128*64;  i += 256) { int n = i>>6,  k = i&63;  W1T[i] = (_Float16)W1[k*128+n]; }
  for (int i = t; i < 128*128; i += 256) { int n = i>>7,  k = i&127; W2T[i] = (_Float16)W2[k*128+n]; }
  for (int i = t; i < 16*128;  i += 256) { int n = i>>7,  k = i&127; W3T[i] = (_Float16)W3[k*64+n]; }
  if (t < 128) { b1s[t] = b1[t]; b2s[t] = b2[t]; }
  if (t < 16)  b3s[t] = b3[t];
  for (int i = t; i < 1024; i += 256) Ts[i] = Tg[i];

  const int wave = t >> 5, lane = t & 31;
  const int b0 = blockIdx.x*TILE_B + wave*16;
  _Float16* Xw   = XH2 + wave*16*128;
  _Float16* H1w  = H1  + wave*16*128;
  float* WLSw  = WLS  + wave*256;
  float* COMBw = COMB + wave*256;
  const float w0 = est_w[0];
  const float al = fminf(fmaxf(alpha_p[0], 0.0f), 1.0f);

  // LS estimates: 16 batches x 8 pilots = 128 items, 4 per lane
#pragma unroll
  for (int q = 0; q < 4; ++q) {
    int i = q*32 + lane;
    int m = i >> 3, p = i & 7;
    int bi = b0 + m; if (bi > B-1) bi = B-1;
    const float2 yv = *(const float2*)(Y  + ((size_t)bi*NFFT + pilot_pos[p])*2);
    const float2 xv = *(const float2*)(Xp + ((size_t)bi*NPIL + p)*2);
    float inv = 1.0f / (xv.x*xv.x + xv.y*xv.y);
    float lr = (yv.x*xv.x + yv.y*xv.y) * inv;
    float li = (yv.y*xv.x - yv.x*xv.y) * inv;
    Xw[m*128 + 2*p]     = (_Float16)lr;
    Xw[m*128 + 2*p + 1] = (_Float16)li;
    WLSw[(m*8+p)*2+0] = lr * w0;
    WLSw[(m*8+p)*2+1] = li * w0;
  }
  // zero-pad input columns 16..63 (24 dwords per row, 16 rows = 384 dwords)
  {
    uint32_t* xz = (uint32_t*)Xw;
#pragma unroll
    for (int q = 0; q < 12; ++q) {
      int i = q*32 + lane;
      int m = i/24, c = i%24;
      xz[m*64 + 8 + c] = 0u;
    }
  }
  __syncthreads();

  // layer1: X(16x64) @ W1(64x128) -> relu -> H1
#pragma unroll
  for (int nt = 0; nt < 8; ++nt) {
    v8f acc = {};
#pragma unroll
    for (int ks = 0; ks < 2; ++ks) {
      v16h a = load_A(Xw, 128, lane, ks*32);
      v16h bf = load_B(W1T, 64, lane, nt*16, ks*32);
      acc = __builtin_amdgcn_wmma_f32_16x16x32_f16(false, a, false, bf, (short)0, acc, false, false);
    }
    store_relu(H1w, 128, lane, nt*16, acc, b1s);
  }
  __syncthreads();

  // layer2: H1 @ W2(128x128) -> relu -> XH2 (reuse)
#pragma unroll
  for (int nt = 0; nt < 8; ++nt) {
    v8f acc = {};
#pragma unroll
    for (int ks = 0; ks < 4; ++ks) {
      v16h a = load_A(H1w, 128, lane, ks*32);
      v16h bf = load_B(W2T, 128, lane, nt*16, ks*32);
      acc = __builtin_amdgcn_wmma_f32_16x16x32_f16(false, a, false, bf, (short)0, acc, false, false);
    }
    store_relu(Xw, 128, lane, nt*16, acc, b2s);
  }
  __syncthreads();

  // layer3: only output cols 0..15 (nn_r, nn_i) -> tanh -> blend with weighted LS
  {
    v8f acc = {};
#pragma unroll
    for (int ks = 0; ks < 4; ++ks) {
      v16h a = load_A(Xw, 128, lane, ks*32);
      v16h bf = load_B(W3T, 128, lane, 0, ks*32);
      acc = __builtin_amdgcn_wmma_f32_16x16x32_f16(false, a, false, bf, (short)0, acc, false, false);
    }
    int g = lane >> 4, n = lane & 15;
    float bb = b3s[n];
    int p = n & 7, ri = n >> 3;   // n<8: real, n>=8: imag
#pragma unroll
    for (int r = 0; r < 8; ++r) {
      int m = r + 8*g;
      float o = tanhf(acc[r] + bb);
      float wv = WLSw[(m*8+p)*2 + ri];
      COMBw[m*16 + n] = al*wv + (1.0f - al)*o;
    }
  }
  __syncthreads();

  // final: H_hat[m][k] = sum_p T[k][p] * comb[m][p] (complex), coalesced float2 stores
#pragma unroll 4
  for (int it = 0; it < 32; ++it) {
    int item = it*32 + lane;
    int m = item >> 6, k = item & 63;
    float orr = 0.0f, oii = 0.0f;
#pragma unroll
    for (int p = 0; p < 8; ++p) {
      float cr = COMBw[m*16 + p];
      float ci = COMBw[m*16 + 8 + p];
      float tr = Ts[(k*8+p)*2+0];
      float ti = Ts[(k*8+p)*2+1];
      orr += tr*cr - ti*ci;
      oii += tr*ci + ti*cr;
    }
    int bi = b0 + m;
    if (bi < B) {
      float2 v; v.x = orr; v.y = oii;
      *(float2*)(out + ((size_t)bi*NFFT + k)*2) = v;
    }
  }
}

extern "C" void kernel_launch(void* const* d_in, const int* in_sizes, int n_in,
                              void* d_out, int out_size, void* d_ws, size_t ws_size,
                              hipStream_t stream) {
  const float* Y     = (const float*)d_in[0];
  const float* Xp    = (const float*)d_in[1];
  const int*   pilot = (const int*)d_in[2];
  const float* W1    = (const float*)d_in[3];
  const float* b1    = (const float*)d_in[4];
  const float* W2    = (const float*)d_in[5];
  const float* b2    = (const float*)d_in[6];
  const float* W3    = (const float*)d_in[7];
  const float* b3    = (const float*)d_in[8];
  const float* est_w = (const float*)d_in[9];
  const float* alpha = (const float*)d_in[10];
  const float* decay = (const float*)d_in[11];
  const float* wlog  = (const float*)d_in[12];
  float* out = (float*)d_out;
  float* T   = (float*)d_ws;   // 1024 floats
  int B = in_sizes[0] / (NFFT*2);

  ce_prep<<<1, 64, 0, stream>>>(pilot, decay, wlog, T);

  size_t smem = (size_t)(128*64 + 128*128 + 16*128)*2   // f16 weights
              + (size_t)(128+128+16+1024)*4             // biases + T
              + (size_t)WAVES*16*128*2*2                // XH2 + H1 (f16)
              + (size_t)WAVES*(16*8*2 + 16*16)*4;       // WLS + COMB (f32)
  int grid = (B + TILE_B - 1) / TILE_B;
  ce_main<<<grid, 256, smem, stream>>>(Y, Xp, pilot, W1, b1, W2, b2, W3, b3,
                                       est_w, alpha, T, out, B);
}
